// MemoryBlock_49881750176207
// MI455X (gfx1250) — matrix operand order repaired and checked by reading, not compiled, not run on hardware
//
#include <hip/hip_runtime.h>

// ---------------- problem constants (from reference setup) ----------------
constexpr int NATTN = 256;
constexpr int HID   = 512;                 // H
constexpr int SEQ   = 16;
constexpr int NT    = 32;
constexpr int NA    = 64;
constexpr int ROWS  = NT * NA;             // 2048 rows per timestep (M)
constexpr int NG    = 4 * HID;             // 2048 gate columns (N)
constexpr int KTOT  = 2 * HID;             // 1024 = [x | h] (K)
constexpr int KSTG  = 64;                  // K per LDS stage
constexpr int NSTG  = KTOT / KSTG;         // 16 stages
constexpr int LDA   = 80;                  // padded LDS row stride (halves), 160B (16B aligned)
constexpr int LDB   = 80;

typedef _Float16 v16h __attribute__((ext_vector_type(16)));
typedef _Float16 h8   __attribute__((ext_vector_type(8)));
typedef float    v8f  __attribute__((ext_vector_type(8)));

// ---------------- CDNA5 async global->LDS path (guarded, with sync fallback) ----
typedef int i128v __attribute__((vector_size(16)));                 // 16-byte payload type
typedef __attribute__((address_space(1))) i128v as1_i128;           // global
typedef __attribute__((address_space(3))) i128v as3_i128;           // LDS

#if defined(__AMDGCN__) && __has_builtin(__builtin_amdgcn_global_load_async_to_lds_b128)
#define HAVE_ASYNC_LDS 1
#else
#define HAVE_ASYNC_LDS 0
#endif

#if HAVE_ASYNC_LDS
// generic LDS pointer low 32 bits == LDS segment offset on amdgcn
#define ASYNC_COPY_B128(g, l)                                                     \
    __builtin_amdgcn_global_load_async_to_lds_b128(                               \
        (as1_i128*)(uintptr_t)(g), (as3_i128*)(unsigned)(uintptr_t)(l), 0, 0)
#if __has_builtin(__builtin_amdgcn_s_wait_asynccnt)
#define WAIT_ASYNC(n) __builtin_amdgcn_s_wait_asynccnt(n)
#else
#define WAIT_ASYNC(n) asm volatile("s_wait_asynccnt " #n)
#endif
#else
#define ASYNC_COPY_B128(g, l) (*(h8*)(l) = *(const h8*)(g))
#define WAIT_ASYNC(n)
#endif

// ---------------- prep kernels ----------------

// Time-major fp16 activations: X[t][row=n*64+a][f] from h_self[:, :, 256:512] | h_inter
__global__ void pack_x_kernel(const float* __restrict__ h_self,
                              const float* __restrict__ h_inter,
                              _Float16* __restrict__ X) {
    int idx = blockIdx.x * blockDim.x + threadIdx.x;
    if (idx >= SEQ * ROWS * HID) return;
    int t   = idx / (ROWS * HID);
    int rem = idx - t * (ROWS * HID);
    int row = rem >> 9;            // /512
    int f   = rem & (HID - 1);
    int n = row >> 6, a = row & 63;
    int bsi = n * SEQ + t;
    float v;
    if (f < NATTN)
        v = h_self[((size_t)bsi * NA + a) * (2 * NATTN) + NATTN + f];
    else
        v = h_inter[((size_t)bsi * NA + a) * (HID - NATTN) + (f - NATTN)];
    X[idx] = (_Float16)v;
}

// Wcat[j][k] fp16: k<512 -> W_ih[j][k], else W_hh[j][k-512]   (B matrix, N-major)
__global__ void pack_w_kernel(const float* __restrict__ W_ih,
                              const float* __restrict__ W_hh,
                              _Float16* __restrict__ Wc) {
    int idx = blockIdx.x * blockDim.x + threadIdx.x;
    if (idx >= NG * KTOT) return;
    int j = idx >> 10;             // /1024
    int k = idx & (KTOT - 1);
    float v = (k < HID) ? W_ih[(size_t)j * HID + k]
                        : W_hh[(size_t)j * HID + (k - HID)];
    Wc[idx] = (_Float16)v;
}

// H = h0*(1-r0) as fp16, C = c0*(1-r0) as fp32
__global__ void init_hc_kernel(const float* __restrict__ hxs,
                               const float* __restrict__ cxs,
                               const float* __restrict__ reset,
                               _Float16* __restrict__ Hb,
                               float* __restrict__ Cb) {
    int idx = blockIdx.x * blockDim.x + threadIdx.x;
    if (idx >= ROWS * HID) return;
    int row = idx >> 9;
    int f   = idx & (HID - 1);
    int n = row >> 6, a = row & 63;
    float keep = 1.0f - reset[n * SEQ + 0];
    size_t src = ((size_t)(n * SEQ) * NA + a) * HID + f;
    Hb[idx] = (_Float16)(hxs[src] * keep);
    Cb[idx] = cxs[src] * keep;
}

// ---------------- WMMA fragment loaders from LDS ----------------
// A fragment (16x32 f16): lane<16: row=lane, K {kk..kk+7, kk+16..kk+23}
//                         lane>=16: row=lane-16, K {kk+8..15, kk+24..31}
__device__ __forceinline__ v16h lds_a_frag(const _Float16* sA, int m0, int kk, int lane) {
    const _Float16* p = sA + (m0 + (lane & 15)) * LDA + kk + ((lane >> 4) << 3);
    union { v16h v; h8 h[2]; } u;
    u.h[0] = *(const h8*)(p);
    u.h[1] = *(const h8*)(p + 16);
    return u.v;
}

// B fragment (32x16 f16) from N-major tile: lane<16: col=lane, K {kk..kk+7, kk+16..kk+23}
//                                           lane>=16: col=lane-16, K {kk+8..15, kk+24..31}
__device__ __forceinline__ v16h lds_b_frag(const _Float16* sB, int n0, int kk, int lane) {
    const _Float16* p = sB + (n0 + (lane & 15)) * LDB + kk + ((lane >> 4) << 3);
    union { v16h v; h8 h[2]; } u;
    u.h[0] = *(const h8*)(p);
    u.h[1] = *(const h8*)(p + 16);
    return u.v;
}

// Issue one stage of async copies: A 64x64 halves (512 x 16B chunks, 2/thread),
// B 128x64 halves (1024 x 16B chunks, 4/thread). 6 async ops per thread total.
__device__ __forceinline__ void stage_copy(const _Float16* __restrict__ Ag, // (blockM row0, ka), ld=HID
                                           const _Float16* __restrict__ Bg, // (blockN col0, k0), ld=KTOT
                                           _Float16* sAbuf, _Float16* sBbuf, int tid) {
    #pragma unroll
    for (int i = 0; i < 2; ++i) {
        int c   = tid + (i << 8);
        int row = c >> 3, col = (c & 7) << 3;
        ASYNC_COPY_B128(Ag + (size_t)row * HID + col, sAbuf + row * LDA + col);
    }
    #pragma unroll
    for (int i = 0; i < 4; ++i) {
        int c   = tid + (i << 8);
        int row = c >> 3, col = (c & 7) << 3;
        ASYNC_COPY_B128(Bg + (size_t)row * KTOT + col, sBbuf + row * LDB + col);
    }
}

// ---------------- fused-K recurrent GEMM: gates = [X_t | H] @ Wcat^T ----------------
// Block: 256 threads = 8 waves as 2(M) x 4(N); wave tile 32x32 -> block tile 64x128.
// K staged through double-buffered LDS with async global->LDS copies.
__global__ __launch_bounds__(256) void lstm_gemm_kernel(
        const _Float16* __restrict__ Xt,    // [ROWS][HID]
        const _Float16* __restrict__ Hb,    // [ROWS][HID]
        const _Float16* __restrict__ Wc,    // [NG][KTOT]
        float* __restrict__ gates) {        // [ROWS][NG]
    __shared__ __align__(16) _Float16 sA[2][64 * LDA];    // 2 x 10 KB
    __shared__ __align__(16) _Float16 sB[2][128 * LDB];   // 2 x 20 KB

    const int tid  = threadIdx.x;
    const int lane = tid & 31;
    const int w    = tid >> 5;
    const int wm   = w >> 2;               // 0..1
    const int wn   = w & 3;                // 0..3
    const int mBlk = blockIdx.x * 64;
    const int nBlk = blockIdx.y * 128;

    // A source for stage starting at k0: X_t for k<512, H for k>=512 (never straddles)
    auto a_src = [&](int k0) -> const _Float16* {
        return (k0 < HID) ? (Xt + (size_t)mBlk * HID + k0)
                          : (Hb + (size_t)mBlk * HID + (k0 - HID));
    };

    v8f acc00 = {}, acc01 = {}, acc10 = {}, acc11 = {};

    // prologue: stage 0
    stage_copy(a_src(0), Wc + (size_t)nBlk * KTOT, sA[0], sB[0], tid);

    for (int s = 0; s < NSTG; ++s) {
        const int cur = s & 1;
        if (s + 1 < NSTG) {
            const int k0 = (s + 1) * KSTG;
            stage_copy(a_src(k0), Wc + (size_t)nBlk * KTOT + k0,
                       sA[1 - cur], sB[1 - cur], tid);
            WAIT_ASYNC(6);                 // current stage (6 older ops) complete
        } else {
            WAIT_ASYNC(0);
        }
        __syncthreads();                   // all waves' stage data visible

        const _Float16* a = sA[cur];
        const _Float16* b = sB[cur];
        #pragma unroll
        for (int kk = 0; kk < KSTG; kk += 32) {
            v16h a0 = lds_a_frag(a, wm * 32,      kk, lane);
            v16h a1 = lds_a_frag(a, wm * 32 + 16, kk, lane);
            v16h b0 = lds_b_frag(b, wn * 32,      kk, lane);
            v16h b1 = lds_b_frag(b, wn * 32 + 16, kk, lane);
            acc00 = __builtin_amdgcn_wmma_f32_16x16x32_f16(false, a0, false, b0, (short)0, acc00, false, false);
            acc01 = __builtin_amdgcn_wmma_f32_16x16x32_f16(false, a0, false, b1, (short)0, acc01, false, false);
            acc10 = __builtin_amdgcn_wmma_f32_16x16x32_f16(false, a1, false, b0, (short)0, acc10, false, false);
            acc11 = __builtin_amdgcn_wmma_f32_16x16x32_f16(false, a1, false, b1, (short)0, acc11, false, false);
        }
        __syncthreads();                   // reads done before buffer is overwritten
    }

    // C/D layout: VGPR r -> M = r + (lane>=16 ? 8 : 0), N = lane&15
    const int rowOff = (lane >> 4) << 3;
    const int colOff = lane & 15;
    const int mBase  = mBlk + wm * 32;
    const int nBase  = nBlk + wn * 32;
    const v8f* accs[2][2] = {{&acc00, &acc01}, {&acc10, &acc11}};
    #pragma unroll
    for (int i = 0; i < 2; ++i)
        #pragma unroll
        for (int j = 0; j < 2; ++j) {
            const v8f acc = *accs[i][j];
            int gm = mBase + i * 16 + rowOff;
            int gn = nBase + j * 16 + colOff;
            #pragma unroll
            for (int r = 0; r < 8; ++r)
                gates[(size_t)(gm + r) * NG + gn] = acc[r];
        }
}

// ---------------- pointwise LSTM cell + reset-for-next-step ----------------
__global__ void lstm_pointwise_kernel(const float* __restrict__ gates,
                                      float* __restrict__ Cb,
                                      _Float16* __restrict__ Hb,
                                      const float* __restrict__ b_ih,
                                      const float* __restrict__ b_hh,
                                      const float* __restrict__ reset,
                                      float* __restrict__ out_h,
                                      float* __restrict__ out_c,
                                      int t) {
    int idx = blockIdx.x * blockDim.x + threadIdx.x;
    if (idx >= ROWS * HID) return;
    int row = idx >> 9;
    int j   = idx & (HID - 1);
    const float* g = gates + (size_t)row * NG;
    float gi = g[j]            + b_ih[j]            + b_hh[j];
    float gf = g[HID + j]      + b_ih[HID + j]      + b_hh[HID + j];
    float gg = g[2 * HID + j]  + b_ih[2 * HID + j]  + b_hh[2 * HID + j];
    float go = g[3 * HID + j]  + b_ih[3 * HID + j]  + b_hh[3 * HID + j];
    float i_ = 1.0f / (1.0f + __expf(-gi));
    float f_ = 1.0f / (1.0f + __expf(-gf));
    float g_ = tanhf(gg);
    float o_ = 1.0f / (1.0f + __expf(-go));
    float c  = f_ * Cb[idx] + i_ * g_;
    float h  = o_ * tanhf(c);

    int n = row >> 6, a = row & 63;
    size_t ob = ((size_t)(n * SEQ + t) * NA + a) * HID + j;
    out_h[ob] = h;
    out_c[ob] = c;

    if (t + 1 < SEQ) {
        float keep = 1.0f - reset[n * SEQ + (t + 1)];
        Cb[idx] = c * keep;
        Hb[idx] = (_Float16)(h * keep);
    }
}

// ---------------- launcher ----------------
extern "C" void kernel_launch(void* const* d_in, const int* in_sizes, int n_in,
                              void* d_out, int out_size, void* d_ws, size_t ws_size,
                              hipStream_t stream) {
    const float* h_self  = (const float*)d_in[0];
    const float* h_inter = (const float*)d_in[1];
    const float* hxs     = (const float*)d_in[2];
    const float* cxs     = (const float*)d_in[3];
    const float* reset   = (const float*)d_in[4];
    const float* W_ih    = (const float*)d_in[5];
    const float* W_hh    = (const float*)d_in[6];
    const float* b_ih    = (const float*)d_in[7];
    const float* b_hh    = (const float*)d_in[8];

    // workspace carve-up (all offsets 256B aligned)
    char* ws = (char*)d_ws;
    size_t off = 0;
    _Float16* X  = (_Float16*)(ws + off); off += (size_t)SEQ * ROWS * HID * 2;   // 32 MB
    _Float16* Wc = (_Float16*)(ws + off); off += (size_t)NG * KTOT * 2;          //  4 MB
    _Float16* Hb = (_Float16*)(ws + off); off += (size_t)ROWS * HID * 2;         //  2 MB
    float*    Cb = (float*)   (ws + off); off += (size_t)ROWS * HID * 4;         //  4 MB
    float*    G  = (float*)   (ws + off); off += (size_t)ROWS * NG * 4;          // 16 MB
    (void)ws_size;

    float* out_h = (float*)d_out;
    float* out_c = out_h + (size_t)SEQ * ROWS * HID;   // hys then cys

    pack_x_kernel<<<(SEQ * ROWS * HID + 255) / 256, 256, 0, stream>>>(h_self, h_inter, X);
    pack_w_kernel<<<(NG * KTOT + 255) / 256, 256, 0, stream>>>(W_ih, W_hh, Wc);
    init_hc_kernel<<<(ROWS * HID + 255) / 256, 256, 0, stream>>>(hxs, cxs, reset, Hb, Cb);

    dim3 ggrid(ROWS / 64, NG / 128);
    for (int t = 0; t < SEQ; ++t) {
        lstm_gemm_kernel<<<ggrid, 256, 0, stream>>>(X + (size_t)t * ROWS * HID, Hb, Wc, G);
        lstm_pointwise_kernel<<<(ROWS * HID + 255) / 256, 256, 0, stream>>>(
            G, Cb, Hb, b_ih, b_hh, reset, out_h, out_c, t);
    }
}
